// HierMPNN_attention_set_67388036874514
// MI455X (gfx1250) — compile-verified, matrix operand order checked by reading, not run on hardware
//
#include <hip/hip_runtime.h>
#include <hip/hip_bf16.h>

// ---------------- CDNA5 WMMA types ----------------
typedef __attribute__((ext_vector_type(16))) __bf16 v16bf;
typedef __attribute__((ext_vector_type(8)))  __bf16 v8bf;
typedef __attribute__((ext_vector_type(2)))  __bf16 v2bf;
typedef __attribute__((ext_vector_type(8)))  float  v8f;

// ---------------- bf16 WMMA GEMM, 64x32 C-block per wave ----------------
// A: row-major bf16 [M, lda], M % 64 == 0, K % 32 == 0.
// Wp: weights pre-packed into WMMA B-fragment order:
//     frag(tn, kc) = 32 lanes x 16 bf16, element addr = ((tn*kch+kc)*32+lane)*16.
// Per CDNA5 ISA 16-bit A layout, lane (m = lane&15, half = lane>>4) holds
// K-runs [k0+half*8 .. +7] and [k0+16+half*8 .. +7] -> two b128 loads.
__global__ __launch_bounds__(32)
void k_gemm_bf16(const __bf16* __restrict__ A, int lda,
                 const __bf16* __restrict__ Wp,
                 const float* __restrict__ bias,
                 const float* __restrict__ bng, const float* __restrict__ bnb,
                 const float* __restrict__ bnm, const float* __restrict__ bnv,
                 const float* __restrict__ addsrc, int ldadd,
                 const float* __restrict__ addscale,
                 float* __restrict__ Cf, __bf16* __restrict__ Cbf, int ldc,
                 int K, int Nout, int do_relu)
{
    const int tm0  = blockIdx.x * 64;
    const int tnB  = blockIdx.y * 2;     // pair of 16-wide N tiles
    const int lane = threadIdx.x;
    const int half = lane >> 4;
    const int l15  = lane & 15;
    const int kch  = K >> 5;

    v8f acc[4][2] = {};
    for (int kc = 0; kc < kch; ++kc) {
        v16bf b0 = *reinterpret_cast<const v16bf*>(
            Wp + (((size_t)tnB * kch + kc) << 9) + (lane << 4));
        v16bf b1 = *reinterpret_cast<const v16bf*>(
            Wp + ((((size_t)tnB + 1) * kch + kc) << 9) + (lane << 4));
#pragma unroll
        for (int mt = 0; mt < 4; ++mt) {
            const __bf16* ap = A + (size_t)(tm0 + mt * 16 + l15) * lda + (kc << 5) + half * 8;
            v8bf lo = *reinterpret_cast<const v8bf*>(ap);
            v8bf hi = *reinterpret_cast<const v8bf*>(ap + 16);
            v16bf a = __builtin_shufflevector(lo, hi,
                        0, 1, 2, 3, 4, 5, 6, 7, 8, 9, 10, 11, 12, 13, 14, 15);
            acc[mt][0] = __builtin_amdgcn_wmma_f32_16x16x32_bf16(
                false, a, false, b0, (short)0, acc[mt][0], false, false);
            acc[mt][1] = __builtin_amdgcn_wmma_f32_16x16x32_bf16(
                false, a, false, b1, (short)0, acc[mt][1], false, false);
        }
    }

#pragma unroll
    for (int mt = 0; mt < 4; ++mt) {
#pragma unroll
        for (int t = 0; t < 2; ++t) {
#pragma unroll
            for (int r = 0; r < 8; ++r) {
                const int row = tm0 + mt * 16 + r + (half << 3);
                const int col = (tnB + t) * 16 + l15;
                if (col < Nout) {
                    float v = acc[mt][t][r];
                    if (bias) v += bias[col];
                    if (bng)  v = (v - bnm[col]) * rsqrtf(bnv[col] + 1e-5f) * bng[col] + bnb[col];
                    if (addsrc) {
                        float s = addscale ? addscale[row] : 1.0f;
                        v += addsrc[(size_t)row * ldadd + col] * s;
                    }
                    if (do_relu) v = fmaxf(v, 0.0f);
                    if (Cf)  Cf[(size_t)row * ldc + col]  = v;
                    if (Cbf) Cbf[(size_t)row * ldc + col] = (__bf16)v;
                }
            }
        }
    }
}

// ---------------- packing ----------------
// Pack f32 weight [rs, cs] (leading dim ldw) into padded [K, N] bf16
// B-fragment layout; OOB -> 0.
__global__ void k_packW(const float* __restrict__ W, int ldw, int rs, int cs,
                        int K, int N, __bf16* __restrict__ Wp)
{
    int i = blockIdx.x * blockDim.x + threadIdx.x;
    if (i >= K * N) return;
    int frag = i >> 9, rem = i & 511;
    int lane = rem >> 4, j = rem & 15;
    int kch = K >> 5;
    int nt = frag / kch, kc = frag - nt * kch;
    int half = lane >> 4, l15 = lane & 15;
    int kk = (kc << 5) + ((j < 8) ? (half * 8 + j) : (16 + half * 8 + (j - 8)));
    int n = nt * 16 + l15;
    float v = (kk < rs && n < cs) ? W[(size_t)kk * ldw + n] : 0.0f;
    Wp[i] = (__bf16)v;
}

// Zero-padded f32 -> row-major bf16 activation copy.
__global__ void k_padbf(const float* __restrict__ src, int ld_src, int rs, int cs,
                        __bf16* __restrict__ dst, int rd, int cd)
{
    int i = blockIdx.x * blockDim.x + threadIdx.x;
    if (i >= rd * cd) return;
    int r = i / cd, c = i % cd;
    dst[i] = (__bf16)((r < rs && c < cs) ? src[(size_t)r * ld_src + c] : 0.0f);
}

__global__ void k_zero(float* p, long n) {
    long i = (long)blockIdx.x * blockDim.x + threadIdx.x;
    if (i < n) p[i] = 0.0f;
}

__global__ void k_zero_bf(__bf16* p, long n) {
    long i = (long)blockIdx.x * blockDim.x + threadIdx.x;
    if (i < n) p[i] = (__bf16)0.0f;
}

__global__ void k_deg(const int* __restrict__ ei, int E, float* __restrict__ deg) {
    int e = blockIdx.x * blockDim.x + threadIdx.x;
    if (e < E) atomicAdd(&deg[ei[E + e]], 1.0f);
}

__global__ void k_invdeg(float* d, int n) {
    int i = blockIdx.x * blockDim.x + threadIdx.x;
    if (i < n) d[i] = 1.0f / fmaxf(d[i], 1.0f);
}

// One wave per edge: msg = h[src] (1x64) @ we[e] (64x64 bf16), scatter-add to agg[dst].
__global__ __launch_bounds__(32)
void k_msg(const float* __restrict__ h, const __bf16* __restrict__ we,
           const int* __restrict__ ei, int E, float* __restrict__ agg)
{
    const int e    = blockIdx.x;
    const int lane = threadIdx.x;
    const int src  = ei[e];
    const int dst  = ei[E + e];
    const __bf16* w = we + (size_t)e * 4096;
    __builtin_prefetch((const void*)(w + lane * 128), 0, 1);  // warm the 8KB matrix
    const float hlo = h[(size_t)src * 64 + lane];
    const float hhi = h[(size_t)src * 64 + 32 + lane];
    float a0 = 0.0f, a1 = 0.0f;
#pragma unroll 8
    for (int k = 0; k < 32; ++k) {
        float hv = __shfl(hlo, k, 32);
        v2bf wv = *reinterpret_cast<const v2bf*>(w + k * 64 + 2 * lane);
        a0 += hv * (float)wv[0];
        a1 += hv * (float)wv[1];
    }
#pragma unroll 8
    for (int k = 0; k < 32; ++k) {
        float hv = __shfl(hhi, k, 32);
        v2bf wv = *reinterpret_cast<const v2bf*>(w + (k + 32) * 64 + 2 * lane);
        a0 += hv * (float)wv[0];
        a1 += hv * (float)wv[1];
    }
    atomicAdd(&agg[(size_t)dst * 64 + 2 * lane],     a0);
    atomicAdd(&agg[(size_t)dst * 64 + 2 * lane + 1], a1);
}

__global__ void k_gru(const float* __restrict__ gx, const float* __restrict__ gh,
                      float* __restrict__ h, __bf16* __restrict__ hb, int n)
{
    int i = blockIdx.x * blockDim.x + threadIdx.x;
    if (i >= n * 64) return;
    int node = i >> 6, j = i & 63;
    const float* px = gx + (size_t)node * 192;
    const float* ph = gh + (size_t)node * 192;
    float r = 1.0f / (1.0f + __expf(-(px[j]      + ph[j])));
    float z = 1.0f / (1.0f + __expf(-(px[64 + j] + ph[64 + j])));
    float nn = tanhf(px[128 + j] + r * ph[128 + j]);
    float hv = (1.0f - z) * nn + z * h[i];
    h[i]  = hv;
    hb[i] = (__bf16)hv;
}

// Multi-head attention over 16-particle sets: 1 block (64 threads) per jet.
__global__ __launch_bounds__(64)
void k_attn(const float* __restrict__ q, const float* __restrict__ Kb,
            const float* __restrict__ Vb, __bf16* __restrict__ ctx)
{
    const int node = blockIdx.x;
    const int t = threadIdx.x;
    const int hh = t >> 4;
    const int dl = t & 15;
    __shared__ float sq[64], ss[64], sa[64];
    sq[t] = q[(size_t)node * 64 + t] * 0.25f;   // 1/sqrt(16)
    __syncthreads();
    float sc = 0.0f;
    const float* kr = Kb + ((size_t)node * 16 + dl) * 64 + hh * 16;
#pragma unroll
    for (int j = 0; j < 16; ++j) sc += sq[hh * 16 + j] * kr[j];
    ss[t] = sc;
    __syncthreads();
    float mx = -3.4e38f;
#pragma unroll
    for (int l = 0; l < 16; ++l) mx = fmaxf(mx, ss[hh * 16 + l]);
    float sum = 0.0f;
#pragma unroll
    for (int l = 0; l < 16; ++l) sum += __expf(ss[hh * 16 + l] - mx);
    sa[t] = __expf(sc - mx) / sum;
    __syncthreads();
    float cv = 0.0f;
#pragma unroll
    for (int l = 0; l < 16; ++l)
        cv += sa[hh * 16 + l] * Vb[((size_t)node * 16 + l) * 64 + hh * 16 + dl];
    ctx[(size_t)node * 64 + hh * 16 + dl] = (__bf16)cv;
}

__global__ void k_lstm(const float* __restrict__ g, float* __restrict__ h,
                       __bf16* __restrict__ hb, float* __restrict__ c, int n)
{
    int i = blockIdx.x * blockDim.x + threadIdx.x;
    if (i >= n * 64) return;
    int b = i >> 6, j = i & 63;
    const float* pg = g + (size_t)b * 256;
    float si = 1.0f / (1.0f + __expf(-pg[j]));
    float sf = 1.0f / (1.0f + __expf(-pg[64 + j]));
    float so = 1.0f / (1.0f + __expf(-pg[192 + j]));
    float c2 = sf * c[i] + si * tanhf(pg[128 + j]);
    float h2 = so * tanhf(c2);
    h[i]  = h2;
    hb[i] = (__bf16)h2;
    c[i]  = c2;
}

// Set2Set pooling: 1 block (64 threads) per graph (8 nodes/graph, d=64).
__global__ __launch_bounds__(64)
void k_s2s_pool(const float* __restrict__ x, const float* __restrict__ h,
                __bf16* __restrict__ qstar)
{
    const int b = blockIdx.x;
    const int t = threadIdx.x;
    __shared__ float sh[64], se[8], sa[8];
    sh[t] = h[(size_t)b * 64 + t];
    __syncthreads();
    if (t < 8) {
        float e = 0.0f;
        const float* xr = x + (size_t)(b * 8 + t) * 64;
        for (int j = 0; j < 64; ++j) e += xr[j] * sh[j];
        se[t] = e;
    }
    __syncthreads();
    if (t == 0) {
        float m = se[0];
        for (int l = 1; l < 8; ++l) m = fmaxf(m, se[l]);
        float s = 0.0f;
        for (int l = 0; l < 8; ++l) { sa[l] = __expf(se[l] - m); s += sa[l]; }
        for (int l = 0; l < 8; ++l) sa[l] /= s;
    }
    __syncthreads();
    float r = 0.0f;
#pragma unroll
    for (int l = 0; l < 8; ++l) r += sa[l] * x[(size_t)(b * 8 + l) * 64 + t];
    qstar[(size_t)b * 128 + t]      = (__bf16)sh[t];
    qstar[(size_t)b * 128 + 64 + t] = (__bf16)r;
}

__global__ void k_logsoftmax(const float* __restrict__ z, float* __restrict__ out, int n)
{
    int i = blockIdx.x * blockDim.x + threadIdx.x;
    if (i >= n) return;
    const float* zi = z + (size_t)i * 4;
    float m = fmaxf(fmaxf(zi[0], zi[1]), fmaxf(zi[2], zi[3]));
    float s = 0.0f;
#pragma unroll
    for (int j = 0; j < 4; ++j) s += __expf(zi[j] - m);
    float ls = logf(s) + m;
#pragma unroll
    for (int j = 0; j < 4; ++j) out[(size_t)i * 4 + j] = zi[j] - ls;
}

// ---------------- Orchestration ----------------
static constexpr int NL = 32768, EL = 32768, NH = 2048, EH = 8192, BB = 256;

extern "C" void kernel_launch(void* const* d_in, const int* in_sizes, int n_in,
                              void* d_out, int out_size, void* d_ws, size_t ws_size,
                              hipStream_t stream)
{
    (void)in_sizes; (void)n_in; (void)out_size; (void)ws_size;
    const float* high_x = (const float*)d_in[0];
    const int*   hei    = (const int*)  d_in[1];
    const float* hea    = (const float*)d_in[2];
    const float* low_x  = (const float*)d_in[4];
    const int*   lei    = (const int*)  d_in[5];
    const float* lea    = (const float*)d_in[6];
#define P(i) ((const float*)d_in[(i)])

    char* wp = (char*)d_ws;
    auto carve = [&](size_t bytes) -> void* {
        void* p = (void*)wp; wp += (bytes + 255) & ~(size_t)255; return p;
    };
    __bf16* we_low  = (__bf16*)carve((size_t)EL * 4096 * 2);
    __bf16* we_high = (__bf16*)carve((size_t)EH * 4096 * 2);
    __bf16* etmp_b  = (__bf16*)carve((size_t)EL * 64 * 2);
    float*  h1      = (float*) carve((size_t)NL * 64 * 4);
    __bf16* h1_b    = (__bf16*)carve((size_t)NL * 64 * 2);
    float*  agg     = (float*) carve((size_t)NL * 64 * 4);
    __bf16* mbuf_b  = (__bf16*)carve((size_t)NL * 64 * 2);
    float*  gx      = (float*) carve((size_t)NL * 192 * 4);
    float*  gh      = (float*) carve((size_t)NL * 192 * 4);
    float*  ideg    = (float*) carve((size_t)NL * 4);
    __bf16* lx_b    = (__bf16*)carve((size_t)NL * 64 * 2);
    __bf16* xq_b    = (__bf16*)carve((size_t)NH * 64 * 2);
    float*  qb      = (float*) carve((size_t)NH * 64 * 4);
    float*  Kb      = (float*) carve((size_t)NL * 64 * 4);
    float*  Vb      = (float*) carve((size_t)NL * 64 * 4);
    __bf16* ctx_b   = (__bf16*)carve((size_t)NH * 64 * 2);
    __bf16* x2_b    = (__bf16*)carve((size_t)NH * 128 * 2);
    float*  h2      = (float*) carve((size_t)NH * 64 * 4);
    __bf16* h2_b    = (__bf16*)carve((size_t)NH * 64 * 2);
    float*  agg2    = (float*) carve((size_t)NH * 64 * 4);
    __bf16* m2_b    = (__bf16*)carve((size_t)NH * 64 * 2);
    float*  gx2     = (float*) carve((size_t)NH * 192 * 4);
    float*  gh2     = (float*) carve((size_t)NH * 192 * 4);
    float*  ideg2   = (float*) carve((size_t)NH * 4);
    float*  hx      = (float*) carve((size_t)NH * 64 * 4);
    __bf16* qstar_b = (__bf16*)carve((size_t)BB * 128 * 2);
    float*  hl      = (float*) carve((size_t)BB * 64 * 4);
    __bf16* hl_b    = (__bf16*)carve((size_t)BB * 64 * 2);
    float*  cl      = (float*) carve((size_t)BB * 64 * 4);
    float*  g1      = (float*) carve((size_t)BB * 256 * 4);
    float*  gat     = (float*) carve((size_t)BB * 256 * 4);
    __bf16* out1_b  = (__bf16*)carve((size_t)BB * 64 * 2);
    float*  logit   = (float*) carve((size_t)BB * 4 * 4);
    // K-padded bf16 activations
    __bf16* lowx_b = (__bf16*)carve((size_t)NL * 32 * 2);
    __bf16* lea_b  = (__bf16*)carve((size_t)EL * 32 * 2);
    __bf16* hix_b  = (__bf16*)carve((size_t)NH * 32 * 2);
    __bf16* hea_b  = (__bf16*)carve((size_t)EH * 32 * 2);
    // packed weights (bf16, fragment layout), padded dims
    auto carveW = [&](int K, int N) { return (__bf16*)carve((size_t)K * N * 2); };
    __bf16* pwm1   = carveW(32, 64);    // conv1.mlp.w  8x64
    __bf16* pwe1l  = carveW(32, 64);    // conv1.e1.w   3x64
    __bf16* pwlin1 = carveW(32, 64);    // conv1.lin.w  8x64
    __bf16* pwe1h  = carveW(32, 64);    // conv2.e1.w   3x64
    __bf16* pwmlp  = carveW(32, 64);    // mlp.w        16x64
    __bf16* pwln   = carveW(32, 64);    // ln.w         16x64
    __bf16* pwe2l  = carveW(64, 4096);
    __bf16* pwe2h  = carveW(64, 4096);
    __bf16* pwrt1  = carveW(64, 64);
    __bf16* pwwi1  = carveW(64, 192);
    __bf16* pwwh1  = carveW(64, 192);
    __bf16* pwrt2  = carveW(64, 64);
    __bf16* pwwi2  = carveW(64, 192);
    __bf16* pwwh2  = carveW(64, 192);
    __bf16* pwq    = carveW(64, 64);
    __bf16* pwk    = carveW(64, 64);
    __bf16* pwv    = carveW(64, 64);
    __bf16* pwo    = carveW(64, 64);
    __bf16* pwm2c  = carveW(128, 64);   // conv2.mlp.w
    __bf16* pwlin2 = carveW(128, 64);   // conv2.lin.w
    __bf16* pwswi  = carveW(128, 256);  // s2s.wi
    __bf16* pwswh  = carveW(64, 256);   // s2s.wh
    __bf16* pwmlp1 = carveW(128, 64);
    __bf16* pwmlp2 = carveW(64, 64);    // mlp2.w 64x4 padded

    auto zero = [&](float* p, long n) {
        k_zero<<<(unsigned)((n + 255) / 256), 256, 0, stream>>>(p, n);
    };
    auto zerob = [&](__bf16* p, long n) {
        k_zero_bf<<<(unsigned)((n + 255) / 256), 256, 0, stream>>>(p, n);
    };
    auto padbf = [&](const float* s, int lds, int rs, int cs, __bf16* d, int rd, int cd) {
        k_padbf<<<(rd * cd + 255) / 256, 256, 0, stream>>>(s, lds, rs, cs, d, rd, cd);
    };
    auto packW = [&](const float* s, int lds, int rs, int cs, int K, int N, __bf16* d) {
        k_packW<<<(K * N + 255) / 256, 256, 0, stream>>>(s, lds, rs, cs, K, N, d);
    };
    auto gemm = [&](const __bf16* A, int lda, const __bf16* W, const float* bias,
                    const float* bg, const float* bb_, const float* bm, const float* bv,
                    const float* adds, int ldadd, const float* ascale,
                    float* Cf, __bf16* Cbf, int ldc, int M, int K, int N, int Nout, int relu) {
        dim3 g(M / 64, N / 32);
        k_gemm_bf16<<<g, 32, 0, stream>>>(A, lda, W, bias, bg, bb_, bm, bv,
                                          adds, ldadd, ascale, Cf, Cbf, ldc, K, Nout, relu);
    };
    const float* Z = nullptr;

    // ---- pack activations and weights ----
    padbf(low_x, 8, NL, 8, lowx_b, NL, 32);
    padbf(lea, 3, EL, 3, lea_b, EL, 32);
    padbf(high_x, 16, NH, 16, hix_b, NH, 32);
    padbf(hea, 3, EH, 3, hea_b, EH, 32);
    packW(P(8), 64, 8, 64, 32, 64, pwm1);
    packW(P(14), 64, 3, 64, 32, 64, pwe1l);
    packW(P(28), 64, 8, 64, 32, 64, pwlin1);
    packW(P(36), 64, 3, 64, 32, 64, pwe1h);
    packW(P(52), 64, 16, 64, 32, 64, pwmlp);
    packW(P(54), 64, 16, 64, 32, 64, pwln);
    packW(P(20), 4096, 64, 4096, 64, 4096, pwe2l);
    packW(P(42), 4096, 64, 4096, 64, 4096, pwe2h);
    packW(P(22), 64, 64, 64, 64, 64, pwrt1);
    packW(P(24), 192, 64, 192, 64, 192, pwwi1);
    packW(P(25), 192, 64, 192, 64, 192, pwwh1);
    packW(P(44), 64, 64, 64, 64, 64, pwrt2);
    packW(P(46), 192, 64, 192, 64, 192, pwwi2);
    packW(P(47), 192, 64, 192, 64, 192, pwwh2);
    packW(P(56), 64, 64, 64, 64, 64, pwq);
    packW(P(58), 64, 64, 64, 64, 64, pwk);
    packW(P(60), 64, 64, 64, 64, 64, pwv);
    packW(P(62), 64, 64, 64, 64, 64, pwo);
    packW(P(30), 64, 128, 64, 128, 64, pwm2c);
    packW(P(50), 64, 128, 64, 128, 64, pwlin2);
    packW(P(64), 256, 128, 256, 128, 256, pwswi);
    packW(P(65), 256, 64, 256, 64, 256, pwswh);
    packW(P(68), 64, 128, 64, 128, 64, pwmlp1);
    packW(P(70), 4, 64, 4, 64, 64, pwmlp2);

    // ===== conv1 (low graph, d=64) =====
    gemm(lowx_b, 32, pwm1, P(9), P(10), P(11), P(12), P(13), Z, 0, Z, h1, h1_b, 64, NL, 32, 64, 64, 1);
    gemm(lea_b, 32, pwe1l, P(15), P(16), P(17), P(18), P(19), Z, 0, Z, nullptr, etmp_b, 64, EL, 32, 64, 64, 1);
    gemm(etmp_b, 64, pwe2l, P(21), Z, Z, Z, Z, Z, 0, Z, nullptr, we_low, 4096, EL, 64, 4096, 4096, 1);
    zero(ideg, NL);
    k_deg<<<(EL + 255) / 256, 256, 0, stream>>>(lei, EL, ideg);
    k_invdeg<<<(NL + 255) / 256, 256, 0, stream>>>(ideg, NL);
    for (int s = 0; s < 3; ++s) {
        zero(agg, (long)NL * 64);
        k_msg<<<EL, 32, 0, stream>>>(h1, we_low, lei, EL, agg);
        gemm(h1_b, 64, pwrt1, P(23), Z, Z, Z, Z, agg, 64, ideg, nullptr, mbuf_b, 64, NL, 64, 64, 64, 1);
        gemm(mbuf_b, 64, pwwi1, P(26), Z, Z, Z, Z, Z, 0, Z, gx, nullptr, 192, NL, 64, 192, 192, 0);
        gemm(h1_b, 64, pwwh1, P(27), Z, Z, Z, Z, Z, 0, Z, gh, nullptr, 192, NL, 64, 192, 192, 0);
        k_gru<<<(NL * 64 + 255) / 256, 256, 0, stream>>>(gx, gh, h1, h1_b, NL);
    }
    gemm(lowx_b, 32, pwlin1, P(29), Z, Z, Z, Z, h1, 64, Z, nullptr, lx_b, 64, NL, 32, 64, 64, 0);

    // ===== attention (dense = lx reshaped [2048,16,64]) =====
    gemm(hix_b, 32, pwmlp, P(53), Z, Z, Z, Z, Z, 0, Z, nullptr, xq_b, 64, NH, 32, 64, 64, 1);
    gemm(xq_b, 64, pwq, P(57), Z, Z, Z, Z, Z, 0, Z, qb, nullptr, 64, NH, 64, 64, 64, 0);
    gemm(lx_b, 64, pwk, P(59), Z, Z, Z, Z, Z, 0, Z, Kb, nullptr, 64, NL, 64, 64, 64, 0);
    gemm(lx_b, 64, pwv, P(61), Z, Z, Z, Z, Z, 0, Z, Vb, nullptr, 64, NL, 64, 64, 64, 0);
    k_attn<<<NH, 64, 0, stream>>>(qb, Kb, Vb, ctx_b);
    gemm(ctx_b, 64, pwo, P(63), Z, Z, Z, Z, Z, 0, Z, nullptr, x2_b + 64, 128, NH, 64, 64, 64, 0);
    gemm(hix_b, 32, pwln, P(55), Z, Z, Z, Z, Z, 0, Z, nullptr, x2_b, 128, NH, 32, 64, 64, 1);

    // ===== conv2 (high graph, d=64, fin=128) =====
    gemm(x2_b, 128, pwm2c, P(31), P(32), P(33), P(34), P(35), Z, 0, Z, h2, h2_b, 64, NH, 128, 64, 64, 1);
    gemm(hea_b, 32, pwe1h, P(37), P(38), P(39), P(40), P(41), Z, 0, Z, nullptr, etmp_b, 64, EH, 32, 64, 64, 1);
    gemm(etmp_b, 64, pwe2h, P(43), Z, Z, Z, Z, Z, 0, Z, nullptr, we_high, 4096, EH, 64, 4096, 4096, 1);
    zero(ideg2, NH);
    k_deg<<<(EH + 255) / 256, 256, 0, stream>>>(hei, EH, ideg2);
    k_invdeg<<<(NH + 255) / 256, 256, 0, stream>>>(ideg2, NH);
    for (int s = 0; s < 3; ++s) {
        zero(agg2, (long)NH * 64);
        k_msg<<<EH, 32, 0, stream>>>(h2, we_high, hei, EH, agg2);
        gemm(h2_b, 64, pwrt2, P(45), Z, Z, Z, Z, agg2, 64, ideg2, nullptr, m2_b, 64, NH, 64, 64, 64, 1);
        gemm(m2_b, 64, pwwi2, P(48), Z, Z, Z, Z, Z, 0, Z, gx2, nullptr, 192, NH, 64, 192, 192, 0);
        gemm(h2_b, 64, pwwh2, P(49), Z, Z, Z, Z, Z, 0, Z, gh2, nullptr, 192, NH, 64, 192, 192, 0);
        k_gru<<<(NH * 64 + 255) / 256, 256, 0, stream>>>(gx2, gh2, h2, h2_b, NH);
    }
    gemm(x2_b, 128, pwlin2, P(51), Z, Z, Z, Z, h2, 64, Z, hx, nullptr, 64, NH, 128, 64, 64, 0);

    // ===== set2set (B=256, 8 nodes/graph) =====
    zerob(qstar_b, (long)BB * 128);
    zero(hl, (long)BB * 64);
    zerob(hl_b, (long)BB * 64);
    zero(cl, (long)BB * 64);
    for (int it = 0; it < 3; ++it) {
        gemm(qstar_b, 128, pwswi, P(66), Z, Z, Z, Z, Z, 0, Z, g1, nullptr, 256, BB, 128, 256, 256, 0);
        gemm(hl_b, 64, pwswh, P(67), Z, Z, Z, Z, g1, 256, Z, gat, nullptr, 256, BB, 64, 256, 256, 0);
        k_lstm<<<(BB * 64 + 255) / 256, 256, 0, stream>>>(gat, hl, hl_b, cl, BB);
        k_s2s_pool<<<BB, 64, 0, stream>>>(hx, hl, qstar_b);
    }

    // ===== head =====
    gemm(qstar_b, 128, pwmlp1, P(69), Z, Z, Z, Z, Z, 0, Z, nullptr, out1_b, 64, BB, 128, 64, 64, 1);
    gemm(out1_b, 64, pwmlp2, P(71), Z, Z, Z, Z, Z, 0, Z, logit, nullptr, 4, BB, 64, 64, 4, 0);
    k_logsoftmax<<<1, 256, 0, stream>>>(logit, (float*)d_out, BB);
#undef P
}
